// Quantize_38809324486818
// MI455X (gfx1250) — matrix operand order repaired and checked by reading, not compiled, not run on hardware
//
#include <hip/hip_runtime.h>
#include <hip/hip_bf16.h>
#include <float.h>

// ---------------------------------------------------------------------------
// VQ (rotation trick) for MI455X / gfx1250, wave32, WMMA f16 16x16x32.
// N=65536 rows, D=512, K=2048 codes.
// Distance GEMM on v_wmma_f32_16x16x32_f16; B tiles staged to LDS via the
// CDNA5 async global->LDS DMA path (ASYNCcnt).
// ---------------------------------------------------------------------------

typedef __attribute__((ext_vector_type(16))) _Float16 v16h;
typedef __attribute__((ext_vector_type(8)))  float    v8f;
typedef __attribute__((vector_size(16)))     int      gv4i;   // matches builtin param

#define N_ROWS   65536
#define DIM      512
#define KCODES   2048
#define NTILES   (KCODES / 16)   // 128 code tiles of 16
#define DSTEPS   (DIM / 32)      // 16 k-steps of 32 per WMMA chain
#define WG       256
#define WAVES    8               // 8 waves * 16 rows = 128 rows per WG
#define MAIN_GRID (N_ROWS / (16 * WAVES))  // 512 workgroups

#if defined(__AMDGCN__) && \
    __has_builtin(__builtin_amdgcn_global_load_async_to_lds_b128) && \
    __has_builtin(__builtin_amdgcn_s_wait_asynccnt)
#define USE_ASYNC_LDS 1
#else
#define USE_ASYNC_LDS 0
#endif

__device__ __forceinline__ float dot4(const float4& a, const float4& b) {
  return a.x * b.x + a.y * b.y + a.z * b.z + a.w * b.w;
}

#if USE_ASYNC_LDS
__device__ __forceinline__ void async_copy16(float4* lds_dst, const float4* gsrc) {
  __builtin_amdgcn_global_load_async_to_lds_b128(
      (__attribute__((address_space(1))) gv4i*)gsrc,
      (__attribute__((address_space(3))) gv4i*)lds_dst,
      /*offset=*/0, /*cpol=*/0);
}
#endif

// ---------------------------------------------------------------------------
// Pack codebook [K, D] f32 -> f16, pre-swizzled into this kernel's B lane layout (contiguous k, same order as A).
// B tile (nt, dstep) is 32(K) x 16(N). Lane L holds n = nt*16 + (L&15),
// k = (L>>4)*16 + e  (e = 0..15), i.e. d = dstep*32 + (L>>4)*16 + e.
// Flat layout: pB[(((nt*DSTEPS + dstep)*32 + lane)*16 + e]
// ---------------------------------------------------------------------------
__global__ void vq_pack_codebook(const float* __restrict__ cb,
                                 _Float16* __restrict__ pB) {
  int t = blockIdx.x * blockDim.x + threadIdx.x;  // 65536 threads total
  int lane  = t & 31;
  int dstep = (t >> 5) & (DSTEPS - 1);
  int ntile = t >> 9;
  int n  = ntile * 16 + (lane & 15);
  int d0 = dstep * 32 + (lane >> 4) * 16;
  const float* src = cb + (size_t)n * DIM + d0;
  _Float16* dst = pB + (((size_t)ntile * DSTEPS + dstep) * 32 + lane) * 16;
#pragma unroll
  for (int e = 0; e < 16; ++e) dst[e] = (_Float16)src[e];
}

// ---------------------------------------------------------------------------
// Exact f32 codebook squared norms (one wave per row).
// ---------------------------------------------------------------------------
__global__ void vq_codebook_norms(const float* __restrict__ cb,
                                  float* __restrict__ cn) {
  int wave = threadIdx.x >> 5, lane = threadIdx.x & 31;
  int row = blockIdx.x * 8 + wave;
  const float* p = cb + (size_t)row * DIM;
  float s = 0.f;
#pragma unroll
  for (int j = 0; j < DIM; j += 32) { float v = p[j + lane]; s += v * v; }
#pragma unroll
  for (int off = 16; off > 0; off >>= 1) s += __shfl_xor(s, off, 32);
  if (lane == 0) cn[row] = s;
}

// 16-WMMA accumulation chain over one 16x16 code tile (K = 512).
__device__ __forceinline__ v8f wmma_tile(const v16h* __restrict__ Bt,
                                         const v16h* __restrict__ A, int lane) {
  v8f acc = {};
#pragma unroll
  for (int i = 0; i < DSTEPS; ++i)
    acc = __builtin_amdgcn_wmma_f32_16x16x32_f16(
        false, A[i], false, Bt[i * 32 + lane], (short)0, acc, false, false);
  return acc;
}

// ---------------------------------------------------------------------------
// Main kernel: WMMA distance GEMM + argmin + rotation trick + loss partials.
// ---------------------------------------------------------------------------
__global__ __launch_bounds__(WG, 1)
void vq_main(const float* __restrict__ x, const float* __restrict__ cb,
             const _Float16* __restrict__ pB, const float* __restrict__ cn,
             float* __restrict__ outEmb, float* __restrict__ outIds,
             float* __restrict__ partials) {
  __shared__ alignas(32) float4 ldsB[2][1024];  // double-buffered 16 KB B tiles
  __shared__ int   sIds[WAVES][16];
  __shared__ float sLoss[WAVES];

  const int tid  = threadIdx.x;
  const int lane = tid & 31;
  const int wave = tid >> 5;
  const int g    = lane >> 4;   // K-half / M-half selector
  const int m    = lane & 15;

  const int rowTile = blockIdx.x * WAVES + wave;
  const int row0    = rowTile * 16;

  // ---- Build A (16 rows x 512 d, f16) in registers, contiguous-k order (B packed to match):
  // lane L, element e -> row m = L&15, k = 16*(L>>4) + e
  v16h A[DSTEPS];
  {
    const float* xr = x + (size_t)(row0 + m) * DIM + g * 16;
#pragma unroll
    for (int i = 0; i < DSTEPS; ++i) {
      const float* p = xr + i * 32;
      float4 a0 = *(const float4*)(p);
      float4 a1 = *(const float4*)(p + 4);
      float4 b0 = *(const float4*)(p + 8);
      float4 b1 = *(const float4*)(p + 12);
      v16h a;
      a[0]  = (_Float16)a0.x; a[1]  = (_Float16)a0.y;
      a[2]  = (_Float16)a0.z; a[3]  = (_Float16)a0.w;
      a[4]  = (_Float16)a1.x; a[5]  = (_Float16)a1.y;
      a[6]  = (_Float16)a1.z; a[7]  = (_Float16)a1.w;
      a[8]  = (_Float16)b0.x; a[9]  = (_Float16)b0.y;
      a[10] = (_Float16)b0.z; a[11] = (_Float16)b0.w;
      a[12] = (_Float16)b1.x; a[13] = (_Float16)b1.y;
      a[14] = (_Float16)b1.z; a[15] = (_Float16)b1.w;
      A[i] = a;
    }
  }

  // Running per-lane argmin. Lane group g tracks rows r + 8*g for r=0..7.
  float rmin[8]; int ridx[8];
#pragma unroll
  for (int r = 0; r < 8; ++r) { rmin[r] = FLT_MAX; ridx[r] = 0x7fffffff; }

  const float4* pB4 = (const float4*)pB;   // 1024 float4 per 16 KB tile

#if USE_ASYNC_LDS
  // ===== CDNA5 async global->LDS double-buffered pipeline (ASYNCcnt) =====
  // Prologue: DMA tile 0 into buffer 0 (64 B per thread = 4 x b128).
#pragma unroll
  for (int k = 0; k < 4; ++k)
    async_copy16(&ldsB[0][tid + k * WG], pB4 + tid + k * WG);

  for (int nt = 0; nt < NTILES; ++nt) {
    __builtin_amdgcn_s_wait_asynccnt(0);   // tile nt landed (this wave's DMA)
    __syncthreads();                       // all waves' DMA landed; prev buf free
    const int next = (nt + 1 < NTILES) ? nt + 1 : nt;
    float4* dstb = ldsB[(nt + 1) & 1];
    const float4* srcb = pB4 + (size_t)next * 1024;
#pragma unroll
    for (int k = 0; k < 4; ++k)            // DMA overlaps this tile's WMMAs
      async_copy16(&dstb[tid + k * WG], srcb + tid + k * WG);

    v8f acc = wmma_tile((const v16h*)ldsB[nt & 1], A, lane);

    const float c = cn[nt * 16 + m];
    const int id = nt * 16 + m;
#pragma unroll
    for (int r = 0; r < 8; ++r) {
      float sc = c - 2.0f * acc[r];
      if (sc < rmin[r]) { rmin[r] = sc; ridx[r] = id; }
    }
  }
  __builtin_amdgcn_s_wait_asynccnt(0);
#else
  // ===== Fallback: register-staged single-buffer pipeline =====
  float4 pf[4];
#pragma unroll
  for (int k = 0; k < 4; ++k) pf[k] = pB4[tid + k * WG];  // tile 0

  for (int nt = 0; nt < NTILES; ++nt) {
    __syncthreads();                       // previous tile fully consumed
#pragma unroll
    for (int k = 0; k < 4; ++k) ldsB[0][tid + k * WG] = pf[k];
    const int next = (nt + 1 < NTILES) ? nt + 1 : nt;
#pragma unroll
    for (int k = 0; k < 4; ++k)            // loads overlap this tile's WMMAs
      pf[k] = pB4[(size_t)next * 1024 + tid + k * WG];
    __syncthreads();                       // tile nt visible in LDS

    v8f acc = wmma_tile((const v16h*)ldsB[0], A, lane);

    const float c = cn[nt * 16 + m];
    const int id = nt * 16 + m;
#pragma unroll
    for (int r = 0; r < 8; ++r) {
      float sc = c - 2.0f * acc[r];
      if (sc < rmin[r]) { rmin[r] = sc; ridx[r] = id; }
    }
  }
#endif

  // ---- Butterfly argmin across the 16-lane group (ties -> lower index,
  //      matching jnp.argmin first-occurrence semantics).
#pragma unroll
  for (int r = 0; r < 8; ++r) {
#pragma unroll
    for (int off = 1; off < 16; off <<= 1) {
      float ov = __shfl_xor(rmin[r], off, 32);
      int   oi = __shfl_xor(ridx[r], off, 32);
      if (ov < rmin[r] || (ov == rmin[r] && oi < ridx[r])) {
        rmin[r] = ov; ridx[r] = oi;
      }
    }
    if (m == 0) {
      int rw = r + 8 * g;                  // row within this wave's 16-row tile
      sIds[wave][rw] = ridx[r];
      outIds[row0 + rw] = (float)ridx[r];
    }
  }
  __syncthreads();

  // ---- Phase 2: rotation trick (exact f32) + loss partial
  float lsum = 0.f;
  for (int rr = 0; rr < 16; ++rr) {
    const int id = sIds[wave][rr];
    const float* xr = x  + (size_t)(row0 + rr) * DIM;
    const float* er = cb + (size_t)id * DIM;
    float4 xv[4], ev[4];
    float sxx = 0.f, see = 0.f, sxe = 0.f;
#pragma unroll
    for (int j = 0; j < 4; ++j) {
      xv[j] = *(const float4*)(xr + j * 128 + lane * 4);
      ev[j] = *(const float4*)(er + j * 128 + lane * 4);
      sxx += dot4(xv[j], xv[j]);
      see += dot4(ev[j], ev[j]);
      sxe += dot4(xv[j], ev[j]);
      float dx0 = xv[j].x - ev[j].x, dx1 = xv[j].y - ev[j].y;
      float dx2 = xv[j].z - ev[j].z, dx3 = xv[j].w - ev[j].w;
      lsum += dx0 * dx0 + dx1 * dx1 + dx2 * dx2 + dx3 * dx3;
    }
#pragma unroll
    for (int off = 16; off > 0; off >>= 1) {   // all lanes end with full sums
      sxx += __shfl_xor(sxx, off, 32);
      see += __shfl_xor(see, off, 32);
      sxe += __shfl_xor(sxe, off, 32);
    }
    // out = scale * ( x - 2*ew*w + 2*eu*q ) = cx * x + ce * e  (closed form)
    const float nx  = sqrtf(sxx), ne = sqrtf(see);
    const float inx = 1.f / (nx + 1e-8f), ine = 1.f / (ne + 1e-8f);
    const float nuq = sqrtf(sxx * inx * inx + 2.f * sxe * inx * ine + see * ine * ine);
    const float idn = 1.f / fmaxf(nuq, 1e-6f);
    const float ew  = (sxx * inx + sxe * ine) * idn;   // x . w
    const float eu  = sxx * inx;                       // x . u
    const float scale = ne / (nx + 1e-6f);
    const float cx = scale * (1.f - 2.f * ew * idn * inx);
    const float ce = scale * ine * (2.f * eu - 2.f * ew * idn);

    float* orow = outEmb + (size_t)(row0 + rr) * DIM;
#pragma unroll
    for (int j = 0; j < 4; ++j) {
      float4 o;
      o.x = cx * xv[j].x + ce * ev[j].x;
      o.y = cx * xv[j].y + ce * ev[j].y;
      o.z = cx * xv[j].z + ce * ev[j].z;
      o.w = cx * xv[j].w + ce * ev[j].w;
      *(float4*)(orow + j * 128 + lane * 4) = o;
    }
  }

  // Deterministic loss reduction: lane butterfly -> per-wave -> per-WG partial.
#pragma unroll
  for (int off = 16; off > 0; off >>= 1) lsum += __shfl_xor(lsum, off, 32);
  if (lane == 0) sLoss[wave] = lsum;
  __syncthreads();
  if (tid == 0) {
    float s = 0.f;
    for (int w = 0; w < WAVES; ++w) s += sLoss[w];
    partials[blockIdx.x] = s;
  }
}

// Deterministic fixed-order final reduction of 512 WG partials.
__global__ void vq_finalize_loss(const float* __restrict__ partials,
                                 float* __restrict__ lossOut) {
  if (blockIdx.x == 0 && threadIdx.x == 0) {
    float s = 0.f;
    for (int i = 0; i < MAIN_GRID; ++i) s += partials[i];
    // loss = mean((emb-x)^2) + 0.25*mean((x-emb)^2) = 1.25 * mean((x-emb)^2)
    lossOut[0] = 1.25f * s * (1.0f / ((float)N_ROWS * (float)DIM));
  }
}

extern "C" void kernel_launch(void* const* d_in, const int* in_sizes, int n_in,
                              void* d_out, int out_size, void* d_ws, size_t ws_size,
                              hipStream_t stream) {
  const float* x  = (const float*)d_in[0];   // [N, D] f32
  const float* cb = (const float*)d_in[1];   // [K, D] f32
  (void)in_sizes; (void)n_in; (void)out_size; (void)ws_size;

  float* out     = (float*)d_out;
  float* outEmb  = out;                                    // [N, D]
  float* outIds  = out + (size_t)N_ROWS * DIM;             // [N] (as float)
  float* lossOut = out + (size_t)N_ROWS * DIM + N_ROWS;    // [1]

  char* ws = (char*)d_ws;
  _Float16* pB     = (_Float16*)ws;                             // 2 MB packed B
  float*    cnorm  = (float*)(ws + (size_t)KCODES * DIM * 2);   // 8 KB
  float*    parts  = cnorm + KCODES;                            // 2 KB

  vq_pack_codebook<<<(NTILES * DSTEPS * 32) / 256, 256, 0, stream>>>(cb, pB);
  vq_codebook_norms<<<KCODES / 8, 256, 0, stream>>>(cb, cnorm);
  vq_main<<<MAIN_GRID, WG, 0, stream>>>(x, cb, pB, cnorm, outEmb, outIds, parts);
  vq_finalize_loss<<<1, 1, 0, stream>>>(parts, lossOut);
}
